// LowrankLearnableHash_83064667504941
// MI455X (gfx1250) — matrix operand order repaired and checked by reading, compiled, not run on hardware
//
#include <hip/hip_runtime.h>
#include <hip/hip_bf16.h>
#include <math.h>

#define RESG   512
#define RANKN  16
#define FDIM   32
#define FRES   64
#define HID    64
#define NI     256
#define NRAYS  4096
#define NPTS   (NRAYS * NI)
#define RADIUSF 1.3f
#define STEPF   0.02f

typedef __attribute__((ext_vector_type(16))) _Float16 v16h;
typedef __attribute__((ext_vector_type(8)))  float    v8f;
typedef __attribute__((ext_vector_type(4)))  int      i4;

#define AS1 __attribute__((address_space(1)))
#define AS3 __attribute__((address_space(3)))

#if defined(__has_builtin)
#if __has_builtin(__builtin_amdgcn_global_load_async_to_lds_b128)
#define USE_ASYNC_LDS 1
#endif
#if __has_builtin(__builtin_amdgcn_s_wait_asynccnt)
#define HAVE_WAIT_ASYNC_BUILTIN 1
#endif
#endif

__device__ __forceinline__ void wait_async0() {
#if defined(HAVE_WAIT_ASYNC_BUILTIN)
    __builtin_amdgcn_s_wait_asynccnt(0);
#else
    asm volatile("s_wait_asynccnt 0x0" ::: "memory");
#endif
}

__device__ __forceinline__ int iclamp(int v, int lo, int hi) {
    return v < lo ? lo : (v > hi ? hi : v);
}

// Bilinear sample of 8 consecutive channels [cb, cb+8) from grid g[RANK][RESG][RESG].
// cx indexes the LAST axis (torch grid_sample convention), align_corners=True, zeros pad.
__device__ __forceinline__ void bilerp8(const float* __restrict__ g,
                                        float cx, float cy, int cb, float out[8]) {
    float px = (cx + 1.0f) * 0.5f * (float)(RESG - 1);
    float py = (cy + 1.0f) * 0.5f * (float)(RESG - 1);
    float fx = floorf(px), fy = floorf(py);
    int ix = (int)fx, iy = (int)fy;
    float wx = px - fx, wy = py - fy;
    float wxa[2] = { (1.0f - wx) * ((ix     >= 0 && ix     < RESG) ? 1.0f : 0.0f),
                     wx          * ((ix + 1 >= 0 && ix + 1 < RESG) ? 1.0f : 0.0f) };
    float wya[2] = { (1.0f - wy) * ((iy     >= 0 && iy     < RESG) ? 1.0f : 0.0f),
                     wy          * ((iy + 1 >= 0 && iy + 1 < RESG) ? 1.0f : 0.0f) };
    int ixa[2] = { iclamp(ix, 0, RESG - 1), iclamp(ix + 1, 0, RESG - 1) };
    int iya[2] = { iclamp(iy, 0, RESG - 1), iclamp(iy + 1, 0, RESG - 1) };
#pragma unroll
    for (int c = 0; c < 8; ++c) out[c] = 0.0f;
#pragma unroll
    for (int by = 0; by < 2; ++by) {
#pragma unroll
        for (int bx = 0; bx < 2; ++bx) {
            float w = wxa[bx] * wya[by];
            const float* gp = g + (size_t)iya[by] * RESG + ixa[bx];
#pragma unroll
            for (int c = 0; c < 8; ++c)
                out[c] = fmaf(w, gp[(size_t)(cb + c) * RESG * RESG], out[c]);
        }
    }
}

// Trilinear sample from F[FDIM][64][64][64]: channels [cb0,cb0+8)->out[0..7],
// [cb1,cb1+8)->out[8..15]. cx indexes the LAST axis.
__device__ __forceinline__ void trilerp16(const float* __restrict__ g,
                                          float cx, float cy, float cz,
                                          int cb0, int cb1, float out[16]) {
    float px = (cx + 1.0f) * 0.5f * (float)(FRES - 1);
    float py = (cy + 1.0f) * 0.5f * (float)(FRES - 1);
    float pz = (cz + 1.0f) * 0.5f * (float)(FRES - 1);
    float fx = floorf(px), fy = floorf(py), fz = floorf(pz);
    int ix = (int)fx, iy = (int)fy, iz = (int)fz;
    float wx = px - fx, wy = py - fy, wz = pz - fz;
    float wxa[2] = { (1.0f - wx) * ((ix     >= 0 && ix     < FRES) ? 1.0f : 0.0f),
                     wx          * ((ix + 1 >= 0 && ix + 1 < FRES) ? 1.0f : 0.0f) };
    float wya[2] = { (1.0f - wy) * ((iy     >= 0 && iy     < FRES) ? 1.0f : 0.0f),
                     wy          * ((iy + 1 >= 0 && iy + 1 < FRES) ? 1.0f : 0.0f) };
    float wza[2] = { (1.0f - wz) * ((iz     >= 0 && iz     < FRES) ? 1.0f : 0.0f),
                     wz          * ((iz + 1 >= 0 && iz + 1 < FRES) ? 1.0f : 0.0f) };
    int ixa[2] = { iclamp(ix, 0, FRES - 1), iclamp(ix + 1, 0, FRES - 1) };
    int iya[2] = { iclamp(iy, 0, FRES - 1), iclamp(iy + 1, 0, FRES - 1) };
    int iza[2] = { iclamp(iz, 0, FRES - 1), iclamp(iz + 1, 0, FRES - 1) };
#pragma unroll
    for (int c = 0; c < 16; ++c) out[c] = 0.0f;
    const size_t CH = (size_t)FRES * FRES * FRES;
#pragma unroll
    for (int bz = 0; bz < 2; ++bz) {
#pragma unroll
        for (int by = 0; by < 2; ++by) {
#pragma unroll
            for (int bx = 0; bx < 2; ++bx) {
                float w = wxa[bx] * wya[by] * wza[bz];
                size_t base = ((size_t)iza[bz] * FRES + iya[by]) * FRES + ixa[bx];
#pragma unroll
                for (int c = 0; c < 8; ++c)
                    out[c]     = fmaf(w, g[(size_t)(cb0 + c) * CH + base], out[c]);
#pragma unroll
                for (int c = 0; c < 8; ++c)
                    out[8 + c] = fmaf(w, g[(size_t)(cb1 + c) * CH + base], out[8 + c]);
            }
        }
    }
}

// ---------------------------------------------------------------------------
// Prep: pack W1/Wc1 into the exact WMMA f16 B-tile register layout (8 tiles x
// 32 lanes x 16 halfs = 8 KB) and build coalesced per-column scalar tables:
// scal[0]=b1, [1]=W2, [2]=bc1, [3..5]=Wc2 rows, [6..8]=Wc1 dir columns.
// ---------------------------------------------------------------------------
__global__ void pack_weights_kernel(const float* __restrict__ W1,
                                    const float* __restrict__ b1v,
                                    const float* __restrict__ W2,
                                    const float* __restrict__ Wc1,
                                    const float* __restrict__ bc1,
                                    const float* __restrict__ Wc2,
                                    _Float16* __restrict__ pk,
                                    float* __restrict__ scal)
{
    int tid  = threadIdx.x;      // 0..255 == tile(0..7) * 32 + lane
    int t    = tid >> 5;
    int lane = tid & 31;
    int n    = lane & 15;
    int grp  = lane >> 4;        // K-half: element e maps to K = grp*16 + e
    _Float16* dst = pk + (size_t)tid * 16;
    if (t < 4) {
        const float* src = W1 + (size_t)(t * 16 + n) * FDIM + grp * 16;
#pragma unroll
        for (int e = 0; e < 16; ++e) dst[e] = (_Float16)src[e];
    } else {
        const float* src = Wc1 + (size_t)((t - 4) * 16 + n) * (FDIM + 3) + grp * 16;
#pragma unroll
        for (int e = 0; e < 16; ++e) dst[e] = (_Float16)src[e];
    }
    if (tid < HID) {
        scal[0 * HID + tid] = b1v[tid];
        scal[1 * HID + tid] = W2[tid];
        scal[2 * HID + tid] = bc1[tid];
        scal[3 * HID + tid] = Wc2[0 * HID + tid];
        scal[4 * HID + tid] = Wc2[1 * HID + tid];
        scal[5 * HID + tid] = Wc2[2 * HID + tid];
        scal[6 * HID + tid] = Wc1[(size_t)tid * (FDIM + 3) + 32];
        scal[7 * HID + tid] = Wc1[(size_t)tid * (FDIM + 3) + 33];
        scal[8 * HID + tid] = Wc1[(size_t)tid * (FDIM + 3) + 34];
    }
}

// ---------------------------------------------------------------------------
// Main: one wave32 per tile of 16 consecutive samples of one ray.
// Lanes l and l+16 both own sample (l&15); each computes half the feature
// channels so feats land directly in the WMMA 16x32 f16 A-matrix layout.
// ---------------------------------------------------------------------------
__global__ __launch_bounds__(256) void nerf_point_kernel(
    const float* __restrict__ rays_o, const float* __restrict__ rays_d,
    const float* __restrict__ Gxy, const float* __restrict__ Gxz,
    const float* __restrict__ Gyz, const float* __restrict__ basis_W,
    const float* __restrict__ Fg,
    const _Float16* __restrict__ pk, const float* __restrict__ scal,
    const float* __restrict__ b2v, const float* __restrict__ bc2,
    float* __restrict__ sigma_out, float* __restrict__ rgb_out)
{
    __shared__ __align__(32) _Float16 ldsW[8 * 32 * 16];   // 8 KB packed B tiles

    // ---- stage packed weights into LDS (async copy, once per block) ----
    {
        int tid = threadIdx.x;
        const _Float16* src = pk + (size_t)tid * 16;       // 32 B per thread
#if defined(USE_ASYNC_LDS)
        __builtin_amdgcn_global_load_async_to_lds_b128(
            (AS1 i4*)src, (AS3 i4*)&ldsW[tid * 16], 0, 0);
        __builtin_amdgcn_global_load_async_to_lds_b128(
            (AS1 i4*)src, (AS3 i4*)&ldsW[tid * 16], 16, 0);
        wait_async0();
#else
        const uint4* s4 = (const uint4*)src;
        uint4* d4 = (uint4*)&ldsW[tid * 16];
        d4[0] = s4[0];
        d4[1] = s4[1];
#endif
        __syncthreads();
    }
    const v16h* ldsBT = (const v16h*)ldsW;

    const int lane = threadIdx.x & 31;
    const int wave = threadIdx.x >> 5;
    const int tile = blockIdx.x * 8 + wave;
    const int ray  = tile >> 4;          // NI/16 = 16 tiles per ray
    const int s0   = (tile & 15) << 4;   // first sample index in tile
    const int m    = lane & 15;          // sample row within tile
    const int grp  = lane >> 4;          // K-half selector
    const int sample = s0 + m;
    const int pt   = ray * NI + sample;

    // ---- ray/AABB setup (wave-uniform) ----
    const float ox = rays_o[3 * ray + 0], oy = rays_o[3 * ray + 1], oz = rays_o[3 * ray + 2];
    const float dx = rays_d[3 * ray + 0], dy = rays_d[3 * ray + 1], dz = rays_d[3 * ray + 2];
    float tn, tf;
    {
        float a0 = (-RADIUSF - ox) / dx, c0 = (RADIUSF - ox) / dx;
        float a1 = (-RADIUSF - oy) / dy, c1 = (RADIUSF - oy) / dy;
        float a2 = (-RADIUSF - oz) / dz, c2 = (RADIUSF - oz) / dz;
        float n0 = fminf(a0, c0), x0 = fmaxf(a0, c0);
        float n1 = fminf(a1, c1), x1 = fmaxf(a1, c1);
        float n2 = fminf(a2, c2), x2 = fmaxf(a2, c2);
        tn = fmaxf(fmaxf(fmaxf(n0, n1), n2), 0.0f);
        tf = fminf(fminf(x0, x1), x2);
    }
    const bool ray_ok = (tn < tf);

    // Whole-tile early out (wave-uniform: EXEC stays all-ones for WMMA below).
    if (!ray_ok || (tn + (float)s0 * STEPF) > tf) {
        if (grp == 0) {
            sigma_out[pt] = 0.0f;
            rgb_out[3 * pt + 0] = 0.0f;
            rgb_out[3 * pt + 1] = 0.0f;
            rgb_out[3 * pt + 2] = 0.0f;
        }
        return;
    }

    // ---- sample position ----
    const float ts  = tn + (float)sample * STEPF;
    const float inr = 1.0f / RADIUSF;
    const float pxc = (ox + ts * dx) * inr;
    const float pyc = (oy + ts * dy) * inr;
    const float pzc = (oz + ts * dz) * inr;

    // ---- tri-plane features: this lane covers rank channels [cb, cb+8) ----
    const int cb = grp * 8;
    float xy[8], xz[8], yz[8];
    bilerp8(Gxy, pxc, pyc, cb, xy);   // coords p[:, (0,1)]
    bilerp8(Gxz, pxc, pzc, cb, xz);   // coords p[:, (0,2)]
    bilerp8(Gyz, pyc, pzc, cb, yz);   // coords p[:, (1,2)]

    float gv0 = 0.0f, gv1 = 0.0f, gv2 = 0.0f;
#pragma unroll
    for (int r = 0; r < 8; ++r) {
        float pr = xy[r] * xz[r] * yz[r];
        gv0 = fmaf(pr, basis_W[0 * RANKN + cb + r], gv0);
        gv1 = fmaf(pr, basis_W[1 * RANKN + cb + r], gv1);
        gv2 = fmaf(pr, basis_W[2 * RANKN + cb + r], gv2);
    }
    gv0 += __shfl_xor(gv0, 16);
    gv1 += __shfl_xor(gv1, 16);
    gv2 += __shfl_xor(gv2, 16);

    // ---- 3D feature volume sample, packed into WMMA A layout ----
    float fe[16];
    trilerp16(Fg, gv0, gv1, gv2, grp * 16, grp * 16 + 8, fe);
    v16h a;
#pragma unroll
    for (int e = 0; e < 16; ++e) a[e] = (_Float16)fe[e];

    const int n = m;

    // ---- density MLP: h = relu(feats @ W1^T + b1) via 4x v_wmma_f32_16x16x32_f16 ----
    v8f hC[4];
#pragma unroll
    for (int t = 0; t < 4; ++t) {
        const int nc = t * 16 + n;
        v16h bt = ldsBT[t * 32 + lane];
        v8f c;
        float bb = scal[0 * HID + nc];
#pragma unroll
        for (int v = 0; v < 8; ++v) c[v] = bb;   // bias folded into accumulator
        hC[t] = __builtin_amdgcn_wmma_f32_16x16x32_f16(false, a, false, bt,
                                                       (short)0, c, false, false);
    }
    // sigma = relu(h) @ W2^T + b2
    float part[8];
#pragma unroll
    for (int v = 0; v < 8; ++v) part[v] = 0.0f;
#pragma unroll
    for (int t = 0; t < 4; ++t) {
        float w2n = scal[1 * HID + t * 16 + n];
#pragma unroll
        for (int v = 0; v < 8; ++v)
            part[v] = fmaf(fmaxf(hC[t][v], 0.0f), w2n, part[v]);
    }
#pragma unroll
    for (int off = 1; off < 16; off <<= 1)
#pragma unroll
        for (int v = 0; v < 8; ++v) part[v] += __shfl_xor(part[v], off);
    if (n < 8) {
        int M  = grp * 8 + n;
        int sm = s0 + M;
        bool mk = (tn + (float)sm * STEPF) <= tf;
        sigma_out[ray * NI + sm] = mk ? (part[n] + b2v[0]) : 0.0f;
    }

    // ---- color MLP: dir is wave-uniform -> dir·Wc1[:,32:35] + bc1 folds into C init ----
    float dl  = rsqrtf(dx * dx + dy * dy + dz * dz);
    float rnx = dx * dl, rny = dy * dl, rnz = dz * dl;
    v8f cC[4];
#pragma unroll
    for (int t = 0; t < 4; ++t) {
        const int nc = t * 16 + n;
        v16h bt = ldsBT[(4 + t) * 32 + lane];
        float cin = scal[2 * HID + nc] + rnx * scal[6 * HID + nc]
                  + rny * scal[7 * HID + nc] + rnz * scal[8 * HID + nc];
        v8f c;
#pragma unroll
        for (int v = 0; v < 8; ++v) c[v] = cin;
        cC[t] = __builtin_amdgcn_wmma_f32_16x16x32_f16(false, a, false, bt,
                                                       (short)0, c, false, false);
    }
    // rgb = sigmoid(relu(hc) @ Wc2^T + bc2)
    float pr[8], pg[8], pb[8];
#pragma unroll
    for (int v = 0; v < 8; ++v) { pr[v] = 0.0f; pg[v] = 0.0f; pb[v] = 0.0f; }
#pragma unroll
    for (int t = 0; t < 4; ++t) {
        const int nc = t * 16 + n;
        float wr = scal[3 * HID + nc], wg = scal[4 * HID + nc], wb = scal[5 * HID + nc];
#pragma unroll
        for (int v = 0; v < 8; ++v) {
            float hv = fmaxf(cC[t][v], 0.0f);
            pr[v] = fmaf(hv, wr, pr[v]);
            pg[v] = fmaf(hv, wg, pg[v]);
            pb[v] = fmaf(hv, wb, pb[v]);
        }
    }
#pragma unroll
    for (int off = 1; off < 16; off <<= 1) {
#pragma unroll
        for (int v = 0; v < 8; ++v) {
            pr[v] += __shfl_xor(pr[v], off);
            pg[v] += __shfl_xor(pg[v], off);
            pb[v] += __shfl_xor(pb[v], off);
        }
    }
    if (n < 8) {
        int M  = grp * 8 + n;
        int p2 = ray * NI + s0 + M;
        float r_ = 1.0f / (1.0f + expf(-(pr[n] + bc2[0])));
        float g_ = 1.0f / (1.0f + expf(-(pg[n] + bc2[1])));
        float b_ = 1.0f / (1.0f + expf(-(pb[n] + bc2[2])));
        rgb_out[3 * p2 + 0] = r_;
        rgb_out[3 * p2 + 1] = g_;
        rgb_out[3 * p2 + 2] = b_;
    }
}

// ---------------------------------------------------------------------------
// Compositing: one wave32 per ray. Lane l owns 8 consecutive samples
// (coalesced 32B sigma / 96B rgb per lane). cumprod of (1-alpha+1e-10) via
// per-lane product + __shfl_up wave scan; butterfly reduce for the outputs.
// ---------------------------------------------------------------------------
__global__ __launch_bounds__(256) void composite_kernel(
    const float* __restrict__ sigma, const float* __restrict__ rgb,
    float* __restrict__ out)
{
    const int lane = threadIdx.x & 31;
    const int wave = threadIdx.x >> 5;
    const int ray  = blockIdx.x * 8 + wave;
    const float* sp = sigma + (size_t)ray * NI + lane * 8;
    const float* cp = rgb + ((size_t)ray * NI + lane * 8) * 3;

    float al[8], q[8];
    float lp = 1.0f;
#pragma unroll
    for (int i = 0; i < 8; ++i) {
        float s  = fmaxf(sp[i], 0.0f);
        float aa = 1.0f - expf(-s * STEPF);
        al[i] = aa;
        q[i]  = 1.0f - aa + 1e-10f;
        lp   *= q[i];
    }
    // inclusive scan of per-lane transmittance products
    float inc = lp;
#pragma unroll
    for (int off = 1; off < 32; off <<= 1) {
        float u = __shfl_up(inc, off);
        inc = (lane >= off) ? inc * u : inc;
    }
    float ex = __shfl_up(inc, 1);
    ex = (lane == 0) ? 1.0f : ex;     // exclusive: T at this lane's first sample

    float T = ex, wsum = 0.0f, ar = 0.0f, ag = 0.0f, ab = 0.0f;
#pragma unroll
    for (int i = 0; i < 8; ++i) {
        float w = al[i] * T;
        ar = fmaf(w, cp[3 * i + 0], ar);
        ag = fmaf(w, cp[3 * i + 1], ag);
        ab = fmaf(w, cp[3 * i + 2], ab);
        wsum += w;
        T *= q[i];
    }
#pragma unroll
    for (int off = 1; off < 32; off <<= 1) {
        ar   += __shfl_xor(ar, off);
        ag   += __shfl_xor(ag, off);
        ab   += __shfl_xor(ab, off);
        wsum += __shfl_xor(wsum, off);
    }
    if (lane == 0) {
        float bg = 1.0f - wsum;
        out[3 * ray + 0] = ar + bg;
        out[3 * ray + 1] = ag + bg;
        out[3 * ray + 2] = ab + bg;
    }
}

extern "C" void kernel_launch(void* const* d_in, const int* in_sizes, int n_in,
                              void* d_out, int out_size, void* d_ws, size_t ws_size,
                              hipStream_t stream) {
    (void)in_sizes; (void)n_in; (void)out_size; (void)ws_size;
    const float* rays_o  = (const float*)d_in[0];
    const float* rays_d  = (const float*)d_in[1];
    const float* Gxy     = (const float*)d_in[2];
    const float* Gxz     = (const float*)d_in[3];
    const float* Gyz     = (const float*)d_in[4];
    const float* basis_W = (const float*)d_in[5];
    const float* Fg      = (const float*)d_in[6];
    const float* W1      = (const float*)d_in[7];
    const float* b1v     = (const float*)d_in[8];
    const float* W2      = (const float*)d_in[9];
    const float* b2v     = (const float*)d_in[10];
    const float* Wc1     = (const float*)d_in[11];
    const float* bc1     = (const float*)d_in[12];
    const float* Wc2     = (const float*)d_in[13];
    const float* bc2     = (const float*)d_in[14];

    // workspace layout
    float*     sigma = (float*)d_ws;                       // NPTS floats   (4 MB)
    float*     rgb   = sigma + (size_t)NPTS;               // 3*NPTS floats (12 MB)
    _Float16*  pk    = (_Float16*)(rgb + (size_t)3 * NPTS);// 8*32*16 halfs (8 KB)
    float*     scal  = (float*)(pk + 8 * 32 * 16);         // 9*HID floats

    pack_weights_kernel<<<1, 256, 0, stream>>>(W1, b1v, W2, Wc1, bc1, Wc2, pk, scal);

    const int ntiles = NRAYS * (NI / 16);                  // 65536 tiles, 1 wave each
    nerf_point_kernel<<<ntiles / 8, 256, 0, stream>>>(
        rays_o, rays_d, Gxy, Gxz, Gyz, basis_W, Fg,
        pk, scal, b2v, bc2, sigma, rgb);

    composite_kernel<<<NRAYS / 8, 256, 0, stream>>>(sigma, rgb, (float*)d_out);
}
